// MinkowskiUNet_39883066311119
// MI455X (gfx1250) — matrix operand order repaired and checked by reading, not compiled, not run on hardware
//
#include <hip/hip_runtime.h>
#include <hip/hip_bf16.h>
#include <stdint.h>

typedef __attribute__((ext_vector_type(16))) _Float16 v16h;
typedef __attribute__((ext_vector_type(8)))  _Float16 v8h;
typedef __attribute__((ext_vector_type(4)))  _Float16 v4h;
typedef __attribute__((ext_vector_type(8)))  float    v8f;

// ---------------------------------------------------------------------------
// Implicit-GEMM conv3d (3x3x3 centered) via v_wmma_f32_16x16x32_f16, fully
// templated so all address math folds to shifts/masks and the K loop unrolls.
// Forward:   out[j] = sum_t in[s*j + t - 1] * w[t]
// Transpose: out[p] = sum_t din[p + t - 1] * w[t], din[2c] = in[c]
// A tile: 16 output voxels x 32 k (k = tap*Cin + cin); per lane the A-layout
// is two contiguous 8-K runs -> one b128 (or two b64) loads per run.
// B tile: weights pre-transposed+padded to [CoutPad][KPAD] -> per lane one
// two 8-K runs in LDS, same k order as A -> two ds_load_b128.
// ---------------------------------------------------------------------------

template <int Cin> struct CinLog2      { static constexpr int v = 5; };
template <>        struct CinLog2<4>   { static constexpr int v = 2; };
template <>        struct CinLog2<8>   { static constexpr int v = 3; };
template <>        struct CinLog2<16>  { static constexpr int v = 4; };

// Load one 8-K run of the A fragment starting at k0 (k0 % 8 == 0).
template <int Gi, int Cin, int stride, int transpose>
__device__ inline v8h load_runA(const _Float16* __restrict__ in,
                                int b, int zo, int yo, int xo, int k0) {
  v8h r = {};
  constexpr int SEG  = (Cin < 8) ? Cin : 8;   // 4 or 8 channels per segment
  constexpr int NSEG = 8 / SEG;               // 1 or 2 taps per run
#pragma unroll
  for (int s = 0; s < NSEG; ++s) {
    int k = k0 + s * SEG;
    int t = k >> CinLog2<Cin>::v;             // tap index
    int c = k & (Cin - 1);                    // channel within tap (seg-aligned)
    int kd = t / 9; int tr = t - kd * 9; int kh = tr / 3; int kw = tr - kh * 3;
    int qz, qy, qx; bool ok = (t < 27);
    if (transpose) {
      qz = zo + kd - 1; qy = yo + kh - 1; qx = xo + kw - 1;
      ok = ok && !((qz | qy | qx) & 1);       // dilated input: even coords only
      qz >>= 1; qy >>= 1; qx >>= 1;
    } else {
      qz = zo * stride + kd - 1; qy = yo * stride + kh - 1; qx = xo * stride + kw - 1;
    }
    ok = ok && qz >= 0 && qz < Gi && qy >= 0 && qy < Gi && qx >= 0 && qx < Gi;
    const _Float16* src = &in[(size_t)(((b * Gi + qz) * Gi + qy) * Gi + qx) * Cin + c];
    if (SEG == 8) {
      v8h v = {};
      if (ok) v = *(const v8h*)src;           // global_load_b128
      r = v;
    } else {
      v4h v = {};
      if (ok) v = *(const v4h*)src;           // global_load_b64
#pragma unroll
      for (int j = 0; j < SEG; ++j) r[s * SEG + j] = v[j];
    }
  }
  return r;
}

template <int Gi, int Go, int Cin, int Cout, int Ktot, int stride, int transpose>
__global__ void conv_wmma(const _Float16* __restrict__ in,
                          const _Float16* __restrict__ wtT,   // [CoutPad][KPAD] f16
                          float* __restrict__ out) {
  constexpr int KCH  = (Ktot + 31) / 32;
  constexpr int KPAD = KCH * 32;
  constexpr int NT   = (Cout + 15) / 16;
  constexpr int CoutPad = NT * 16;
  constexpr int Mtot = 2 * Go * Go * Go;       // B = 2
  constexpr int totalTiles = (Mtot / 16) * NT;
  constexpr int LG = (Go == 64) ? 6 : ((Go == 32) ? 5 : 4);

  __shared__ _Float16 wlds[CoutPad * KPAD];
  {
    const uint32_t* src = (const uint32_t*)wtT;
    uint32_t* dst = (uint32_t*)wlds;
    constexpr int nW = (CoutPad * KPAD) / 2;
    for (int i = threadIdx.x; i < nW; i += blockDim.x) dst[i] = src[i];
  }
  __syncthreads();

  int tile = blockIdx.x * 4 + (threadIdx.x >> 5);
  if (tile >= totalTiles) return;
  int lane  = threadIdx.x & 31;
  int ntile = (NT == 1) ? 0 : (tile & (NT - 1));
  int mtile = (NT == 1) ? tile : (tile >> 1);

  int mrow = mtile * 16 + (lane & 15);         // A-fragment row of this lane
  int b  = mrow >> (3 * LG);
  int zo = (mrow >> (2 * LG)) & (Go - 1);
  int yo = (mrow >> LG) & (Go - 1);
  int xo = mrow & (Go - 1);

  // prefetch input near this tile's center tap
  {
    int pz = transpose ? (zo >> 1) : zo * stride;
    int py = transpose ? (yo >> 1) : yo * stride;
    int px = transpose ? (xo >> 1) : xo * stride;
    pz = pz >= Gi ? Gi - 1 : pz; py = py >= Gi ? Gi - 1 : py; px = px >= Gi ? Gi - 1 : px;
    __builtin_prefetch(&in[(size_t)(((b * Gi + pz) * Gi + py) * Gi + px) * Cin], 0, 1);
  }

  int khA = (lane >> 4) << 3;                  // 0 or 8
  int khB = (lane >> 4) << 3;                  // 0 or 8
  const _Float16* wrow = &wlds[(ntile * 16 + (lane & 15)) * KPAD];
  v8f acc = {};

#pragma unroll
  for (int kc = 0; kc < KCH; ++kc) {
    const int kbase = kc * 32;
    v8h a0 = load_runA<Gi, Cin, stride, transpose>(in, b, zo, yo, xo, kbase + khA);
    v8h a1 = load_runA<Gi, Cin, stride, transpose>(in, b, zo, yo, xo, kbase + khA + 16);
    v8h b0 = *(const v8h*)&wrow[kbase + khB];        // ds_load_b128
    v8h b1 = *(const v8h*)&wrow[kbase + khB + 16];   // ds_load_b128
    v16h af, bf;
#pragma unroll
    for (int j = 0; j < 8; ++j) {
      af[j] = a0[j]; af[8 + j] = a1[j];
      bf[j] = b0[j]; bf[8 + j] = b1[j];
    }
    acc = __builtin_amdgcn_wmma_f32_16x16x32_f16(false, af, false, bf,
                                                 (short)0, acc, false, false);
  }

  // D layout: vgpr r -> M = r (+8 for hi lanes), N = lane&15
  int ncol = ntile * 16 + (lane & 15);
#pragma unroll
  for (int r = 0; r < 8; ++r) {
    int m = mtile * 16 + ((lane >> 4) << 3) + r;
    if ((Cout % 16 == 0) || (lane & 15) < Cout)
      out[(size_t)m * Cout + ncol] = acc[r];
  }
}

// ---------------------------------------------------------------------------
// Final 1x1x1 projection: [M,32(cat,padded)] x [32,16] WMMA GEMM, masked out
// ---------------------------------------------------------------------------
__global__ void gemm1x1_wmma(const _Float16* __restrict__ in, const _Float16* __restrict__ wtT,
                             const unsigned char* __restrict__ mask, float* __restrict__ out,
                             int M) {
  int tile = blockIdx.x * 4 + (threadIdx.x >> 5);
  int Mtiles = M >> 4;
  if (tile >= Mtiles) return;
  int lane = threadIdx.x & 31;
  int mrow = tile * 16 + (lane & 15);
  int khA = (lane >> 4) << 3, khB = (lane >> 4) << 3;
  const _Float16* arow = &in[(size_t)mrow * 32];
  v8h a0 = *(const v8h*)&arow[khA];
  v8h a1 = *(const v8h*)&arow[khA + 16];
  const _Float16* wrow = &wtT[(lane & 15) * 32 + khB];
  v8h b0 = *(const v8h*)&wrow[0];
  v8h b1 = *(const v8h*)&wrow[16];
  v16h af, bf;
#pragma unroll
  for (int j = 0; j < 8; ++j) {
    af[j] = a0[j]; af[8 + j] = a1[j];
    bf[j] = b0[j]; bf[8 + j] = b1[j];
  }
  v8f acc = {};
  acc = __builtin_amdgcn_wmma_f32_16x16x32_f16(false, af, false, bf, (short)0, acc, false, false);
#pragma unroll
  for (int r = 0; r < 8; ++r) {
    int m = tile * 16 + ((lane >> 4) << 3) + r;
    float mf = mask[m] ? 1.f : 0.f;
    out[(size_t)m * 16 + (lane & 15)] = acc[r] * mf;
  }
}

// ---------------------------------------------------------------------------
// Masked BatchNorm: deterministic two-stage reduction (64 partials -> combine)
// ---------------------------------------------------------------------------
template <int C>
__global__ void bn_reduce(const float* __restrict__ h, const unsigned char* __restrict__ mask,
                          float* __restrict__ partial, int M) {
  constexpr int NT = 256;
  constexpr int NG = NT / C;
  __shared__ float ls[NT], lss[NT], lcnt[NT];
  int t = threadIdx.x;
  int c = t % C, g = t / C;
  int gg = blockIdx.x * NG + g;
  int stride = gridDim.x * NG;
  float s = 0.f, ss = 0.f, cnt = 0.f;
  for (int r = gg; r < M; r += stride) {
    if (mask[r]) {
      float v = h[(size_t)r * C + c];
      s += v; ss += v * v;
      if (c == 0) cnt += 1.f;
    }
  }
  ls[t] = s; lss[t] = ss; lcnt[t] = cnt;
  __syncthreads();
  for (int gh = NG / 2; gh >= 1; gh >>= 1) {
    if (g < gh) { ls[t] += ls[t + gh * C]; lss[t] += lss[t + gh * C]; lcnt[t] += lcnt[t + gh * C]; }
    __syncthreads();
  }
  if (g == 0) {
    float* o = partial + blockIdx.x * (2 * C + 1);
    o[c] = ls[t]; o[C + c] = lss[t];
    if (c == 0) o[2 * C] = lcnt[t];
  }
}

template <int C>
__global__ void bn_combine(const float* __restrict__ partial, float* __restrict__ stats, int nblk) {
  int t = threadIdx.x;
  if (t >= 2 * C + 1) return;
  float s = 0.f;
  for (int b = 0; b < nblk; ++b) s += partial[b * (2 * C + 1) + t];
  stats[t] = s;
}

// Apply BN (+mask), emit relu f16 activations and/or f16 concat-buffer slices.
template <int C>
__global__ void bn_apply(const float* __restrict__ h, const unsigned char* __restrict__ mask,
                         const float* __restrict__ stats, const float* __restrict__ gamma,
                         const float* __restrict__ beta, _Float16* __restrict__ reluOut,
                         _Float16* __restrict__ catOut, int catOff, int preReluCat,
                         int zeroTail, int M) {
  int r = blockIdx.x * blockDim.x + threadIdx.x;
  if (r >= M) return;
  float mf = mask[r] ? 1.f : 0.f;
  float cnt = stats[2 * C]; cnt = cnt < 1.f ? 1.f : cnt;
#pragma unroll
  for (int c = 0; c < C; ++c) {
    float mean = stats[c] / cnt;
    float var = stats[C + c] / cnt - mean * mean;
    var = var < 0.f ? 0.f : var;
    float inv = rsqrtf(var + 1e-5f);
    float y = ((h[(size_t)r * C + c] - mean) * inv * gamma[c] + beta[c]) * mf;
    float yr = y > 0.f ? y : 0.f;
    if (reluOut) reluOut[(size_t)r * C + c] = (_Float16)yr;
    if (catOut)  catOut[(size_t)r * 32 + catOff + c] = (_Float16)(preReluCat ? y : yr);
  }
  if (catOut && zeroTail) {
    for (int c = catOff + C; c < 32; ++c) catOut[(size_t)r * 32 + c] = (_Float16)0.f;
  }
}

// ---------------------------------------------------------------------------
// Prep kernels
// ---------------------------------------------------------------------------
__global__ void prep_x(const float* __restrict__ x, const unsigned char* __restrict__ m,
                       _Float16* __restrict__ a0, int n) {
  int i = blockIdx.x * blockDim.x + threadIdx.x;
  if (i >= n) return;
  int r = i >> 2;                                  // Cin = 4
  a0[i] = (_Float16)(m[r] ? x[i] : 0.f);
}

__global__ void downsample_mask(const unsigned char* __restrict__ src,
                                unsigned char* __restrict__ dst, int Gc, int Bn) {
  int i = blockIdx.x * blockDim.x + threadIdx.x;
  int tot = Bn * Gc * Gc * Gc;
  if (i >= tot) return;
  int Gf = Gc * 2;
  int x = i % Gc; int t = i / Gc;
  int y = t % Gc; t /= Gc;
  int z = t % Gc; int b = t / Gc;
  unsigned char v = 0;
  for (int dz = 0; dz < 2; ++dz)
    for (int dy = 0; dy < 2; ++dy)
      for (int dx = 0; dx < 2; ++dx)
        v |= src[((b * Gf + (2 * z + dz)) * Gf + (2 * y + dy)) * Gf + (2 * x + dx)];
  dst[i] = v ? 1 : 0;
}

// src: [Ktot][Cout] f32 (DHWIO flattened) -> dst: [CoutPad][KPAD] f16, zero pad
__global__ void cvt_wT(const float* __restrict__ src, _Float16* __restrict__ dst,
                       int Ktot, int Cout, int KPAD, int CoutPad) {
  int i = blockIdx.x * blockDim.x + threadIdx.x;
  int tot = CoutPad * KPAD;
  if (i >= tot) return;
  int n = i / KPAD, k = i - n * KPAD;
  dst[i] = (_Float16)((n < Cout && k < Ktot) ? src[k * Cout + n] : 0.f);
}

// ---------------------------------------------------------------------------
extern "C" void kernel_launch(void* const* d_in, const int* in_sizes, int n_in,
                              void* d_out, int out_size, void* d_ws, size_t ws_size,
                              hipStream_t stream) {
  const float* x  = (const float*)d_in[0];
  const unsigned char* m1 = (const unsigned char*)d_in[1];
  const float *w1 = (const float*)d_in[2],  *g1 = (const float*)d_in[3],  *b1 = (const float*)d_in[4];
  const float *w2 = (const float*)d_in[5],  *g2 = (const float*)d_in[6],  *b2 = (const float*)d_in[7];
  const float *w3 = (const float*)d_in[8],  *g3 = (const float*)d_in[9],  *b3 = (const float*)d_in[10];
  const float *w3t = (const float*)d_in[11], *g3t = (const float*)d_in[12], *b3t = (const float*)d_in[13];
  const float *w2t = (const float*)d_in[14], *g2t = (const float*)d_in[15], *b2t = (const float*)d_in[16];
  const float *w1x1 = (const float*)d_in[17];
  float* out = (float*)d_out;
  (void)in_sizes; (void)n_in; (void)out_size; (void)ws_size;

  const int B = 2;
  const int N1 = B * 64 * 64 * 64;    // 524288 voxels @ 64^3
  const int N2 = B * 32 * 32 * 32;    // 65536  @ 32^3
  const int N4 = B * 16 * 16 * 16;    // 8192   @ 16^3

  char* ws = (char*)d_ws;
  size_t off = 0;
  auto alloc = [&](size_t bytes) -> void* {
    void* p = ws + off;
    off = (off + bytes + 255) & ~(size_t)255;
    return p;
  };
  unsigned char* m2 = (unsigned char*)alloc(N2);
  unsigned char* m4 = (unsigned char*)alloc(N4);
  _Float16* wh1   = (_Float16*)alloc(16 * 128 * 2);   // conv1:  [16][128]
  _Float16* wh2   = (_Float16*)alloc(16 * 224 * 2);   // conv2:  [16][224]
  _Float16* wh3   = (_Float16*)alloc(32 * 448 * 2);   // conv3:  [32][448]
  _Float16* wh3t  = (_Float16*)alloc(16 * 864 * 2);   // convT3: [16][864]
  _Float16* wh2t  = (_Float16*)alloc(16 * 864 * 2);   // convT2: [16][864]
  _Float16* wh1x1 = (_Float16*)alloc(16 * 32 * 2);    // 1x1:    [16][32]
  float* partial  = (float*)alloc(64 * 65 * 4);
  float* st1  = (float*)alloc(17 * 4);
  float* st2  = (float*)alloc(33 * 4);
  float* st3  = (float*)alloc(65 * 4);
  float* st3t = (float*)alloc(33 * 4);
  float* st2t = (float*)alloc(33 * 4);
  _Float16* a0   = (_Float16*)alloc((size_t)N1 * 4 * 2);
  float*    c1   = (float*)   alloc((size_t)N1 * 8 * 4);
  _Float16* a1   = (_Float16*)alloc((size_t)N1 * 8 * 2);
  _Float16* cat1 = (_Float16*)alloc((size_t)N1 * 32 * 2);
  float*    c2   = (float*)   alloc((size_t)N2 * 16 * 4);
  _Float16* a2   = (_Float16*)alloc((size_t)N2 * 16 * 2);
  _Float16* cat2 = (_Float16*)alloc((size_t)N2 * 32 * 2);
  float*    c3   = (float*)   alloc((size_t)N4 * 32 * 4);
  _Float16* a3   = (_Float16*)alloc((size_t)N4 * 32 * 2);
  float*    c3t  = (float*)   alloc((size_t)N2 * 16 * 4);
  float*    c2t  = (float*)   alloc((size_t)N1 * 16 * 4);

  // -- prep: weights -> transposed/padded f16, masks, masked input --
  cvt_wT<<<(16 * 128 + 255) / 256, 256, 0, stream>>>(w1, wh1, 108, 8, 128, 16);
  cvt_wT<<<(16 * 224 + 255) / 256, 256, 0, stream>>>(w2, wh2, 216, 16, 224, 16);
  cvt_wT<<<(32 * 448 + 255) / 256, 256, 0, stream>>>(w3, wh3, 432, 32, 448, 32);
  cvt_wT<<<(16 * 864 + 255) / 256, 256, 0, stream>>>(w3t, wh3t, 864, 16, 864, 16);
  cvt_wT<<<(16 * 864 + 255) / 256, 256, 0, stream>>>(w2t, wh2t, 864, 16, 864, 16);
  cvt_wT<<<(16 * 32 + 255) / 256, 256, 0, stream>>>(w1x1, wh1x1, 24, 16, 32, 16);
  downsample_mask<<<(N2 + 255) / 256, 256, 0, stream>>>(m1, m2, 32, B);
  downsample_mask<<<(N4 + 255) / 256, 256, 0, stream>>>(m2, m4, 16, B);
  prep_x<<<(N1 * 4 + 255) / 256, 256, 0, stream>>>(x, m1, a0, N1 * 4);

  // -- block1: conv(4->8, s1) + BN (pre-relu s1 -> cat1[16:24], zero tail) --
  conv_wmma<64, 64, 4, 8, 108, 1, 0><<<(N1 / 16 + 3) / 4, 128, 0, stream>>>(a0, wh1, c1);
  bn_reduce<8><<<64, 256, 0, stream>>>(c1, m1, partial, N1);
  bn_combine<8><<<1, 96, 0, stream>>>(partial, st1, 64);
  bn_apply<8><<<(N1 + 255) / 256, 256, 0, stream>>>(c1, m1, st1, g1, b1, a1, cat1, 16, 1, 1, N1);

  // -- block2: conv(8->16, s2) + BN (pre-relu s2 -> cat2[16:32]) --
  conv_wmma<64, 32, 8, 16, 216, 2, 0><<<(N2 / 16 + 3) / 4, 128, 0, stream>>>(a1, wh2, c2);
  bn_reduce<16><<<64, 256, 0, stream>>>(c2, m2, partial, N2);
  bn_combine<16><<<1, 96, 0, stream>>>(partial, st2, 64);
  bn_apply<16><<<(N2 + 255) / 256, 256, 0, stream>>>(c2, m2, st2, g2, b2, a2, cat2, 16, 1, 0, N2);

  // -- block3: conv(16->32, s2) + BN --
  conv_wmma<32, 16, 16, 32, 432, 2, 0><<<(N4 / 16 * 2 + 3) / 4, 128, 0, stream>>>(a2, wh3, c3);
  bn_reduce<32><<<64, 256, 0, stream>>>(c3, m4, partial, N4);
  bn_combine<32><<<1, 96, 0, stream>>>(partial, st3, 64);
  bn_apply<32><<<(N4 + 255) / 256, 256, 0, stream>>>(c3, m4, st3, g3, b3, a3,
                                                     (_Float16*)nullptr, 0, 0, 0, N4);

  // -- block3_tr: convT(32->16) to 32^3 + BN + relu -> cat2[0:16] --
  conv_wmma<16, 32, 32, 16, 864, 1, 1><<<(N2 / 16 + 3) / 4, 128, 0, stream>>>(a3, wh3t, c3t);
  bn_reduce<16><<<64, 256, 0, stream>>>(c3t, m2, partial, N2);
  bn_combine<16><<<1, 96, 0, stream>>>(partial, st3t, 64);
  bn_apply<16><<<(N2 + 255) / 256, 256, 0, stream>>>(c3t, m2, st3t, g3t, b3t,
                                                     (_Float16*)nullptr, cat2, 0, 0, 0, N2);

  // -- block2_tr: convT(32->16) on cat2 to 64^3 + BN + relu -> cat1[0:16] --
  conv_wmma<32, 64, 32, 16, 864, 1, 1><<<(N1 / 16 + 3) / 4, 128, 0, stream>>>(cat2, wh2t, c2t);
  bn_reduce<16><<<64, 256, 0, stream>>>(c2t, m1, partial, N1);
  bn_combine<16><<<1, 96, 0, stream>>>(partial, st2t, 64);
  bn_apply<16><<<(N1 + 255) / 256, 256, 0, stream>>>(c2t, m1, st2t, g2t, b2t,
                                                     (_Float16*)nullptr, cat1, 0, 0, 0, N1);

  // -- conv1_tr: 1x1x1 projection [N1,32(pad)] x [32,16], masked output --
  gemm1x1_wmma<<<(N1 / 16 + 3) / 4, 128, 0, stream>>>(cat1, wh1x1, m1, out, N1);
}
